// EncoderDecoderLSTM_69320772158149
// MI455X (gfx1250) — matrix operand order repaired and checked by reading, not compiled, not run on hardware
//
#include <hip/hip_runtime.h>
#include <hip/hip_bf16.h>
#include <cstdint>

// ---------------------------------------------------------------------------
// EncoderDecoderLSTM on gfx1250 (MI455X).
//
// Every LSTM step is one fused kernel:
//   gates = [x|h] @ Wpack^T  (bf16 WMMA 16x16x32, f32 accum)
//   + bias + rank-1 input term, then sigmoid/tanh cell update in epilogue.
// Weights prepacked to bf16, N-major, gate-interleaved (n' = unit*4 + gate).
// h state kept in bf16 (WMMA operand format), c state in f32.
// Global->LDS staging uses gfx1250 async copies (GLOBAL_LOAD_ASYNC_TO_LDS_B128,
// ASYNCcnt) with LDS double-buffering to hide memory latency behind WMMA.
// ---------------------------------------------------------------------------

typedef __attribute__((ext_vector_type(16))) __bf16 v16bf;
typedef __attribute__((ext_vector_type(8)))  float  v8f;

#define B_TOT   1024
#define HID     256
#define G4      1024      // 4*HID
#define LOOKB   336
#define HORIZ   168
#define TM      128       // batch tile
#define TN      64        // n' tile (== 16 hidden units x 4 gates)
#define TK      64        // K staging chunk

// ---------------------------------------------------------------------------
// gfx1250 async global->LDS copy (16B per lane), tracked by ASYNCcnt.
// LDS operand = low 32 bits of the flat shared address (LDS byte offset).
// ---------------------------------------------------------------------------
__device__ __forceinline__ void async_cp16(void* lds, const void* g) {
  asm volatile("global_load_async_to_lds_b128 %0, %1, off"
               :: "v"((uint32_t)(uint64_t)lds), "v"((uint64_t)g)
               : "memory");
}
__device__ __forceinline__ void wait_async0() {
  asm volatile("s_wait_asynccnt 0x0" ::: "memory");
}

// ---------------------------------------------------------------------------
// Weight pack: dst[n'][k] bf16, n' = unit*4+gate (from source row r=gate*256+unit)
// k < Kih comes from Wih (stride Kih), else from Whh (stride K-Kih).
// ---------------------------------------------------------------------------
__global__ void pack_weights(__bf16* __restrict__ dst,
                             const float* __restrict__ Wih,
                             const float* __restrict__ Whh,
                             int Kih, int K) {
  int idx = blockIdx.x * 256 + threadIdx.x;
  if (idx >= G4 * K) return;
  int n = idx / K, k = idx % K;
  int gate = n & 3, unit = n >> 2;
  int r = gate * HID + unit;
  float v = (k < Kih) ? Wih[r * Kih + k] : Whh[r * (K - Kih) + (k - Kih)];
  dst[idx] = (__bf16)v;
}

// Reorder a 4H vector (bias or Wih with input size 1) into gate-interleaved order.
__global__ void pack_vec(float* __restrict__ dst, const float* __restrict__ src) {
  int n = blockIdx.x * 256 + threadIdx.x;
  if (n < G4) { int gate = n & 3, unit = n >> 2; dst[n] = src[gate * HID + unit]; }
}

// Zero the initial read-side state buffers.
__global__ void zero_state(__bf16* h0, __bf16* h1, float* c0, float* c1) {
  int i = blockIdx.x * 256 + threadIdx.x;
  if (i < B_TOT * HID) {
    h0[i] = (__bf16)0.0f; h1[i] = (__bf16)0.0f; c0[i] = 0.0f; c1[i] = 0.0f;
  }
}

// predbuf[b] = x[b][LOOKB-1]  (decoder seed = last observed value)
__global__ void seed_pred(float* __restrict__ predbuf, const float* __restrict__ x) {
  int b = blockIdx.x * 256 + threadIdx.x;
  if (b < B_TOT) predbuf[b] = x[b * LOOKB + (LOOKB - 1)];
}

// pred = h1 @ fcW^T + fcb ; feeds decoder input AND d_out column t.
__global__ void fc_pred(const __bf16* __restrict__ h1,
                        const float* __restrict__ fcW,
                        const float* __restrict__ fcb,
                        float* __restrict__ predbuf,
                        float* __restrict__ out, int t) {
  int b = blockIdx.x * 256 + threadIdx.x;
  if (b >= B_TOT) return;
  const __bf16* row = h1 + b * HID;
  float s = 0.0f;
#pragma unroll 8
  for (int k = 0; k < HID; ++k) s += (float)row[k] * fcW[k];
  float p = s + fcb[0];
  predbuf[b] = p;
  out[b * HORIZ + t] = p;
}

// ---------------------------------------------------------------------------
// Async staging of one 64-wide K chunk: A 128x64 bf16 (4x16B/thread),
// B 64x64 bf16 (2x16B/thread). 6 async ops per thread per chunk.
// ---------------------------------------------------------------------------
__device__ __forceinline__ void stage_chunk(
    const __bf16* __restrict__ x0, const __bf16* __restrict__ x1,
    const __bf16* __restrict__ W, int bb, int nb, int K, int kc0, int tid,
    __bf16 (*As)[72], __bf16 (*Bs)[80]) {
  const __bf16* xsrc = (kc0 < HID) ? x0 : x1;
  const int kl = kc0 & (HID - 1);
#pragma unroll
  for (int it = 0; it < 4; ++it) {
    int chunk = tid + it * 256;
    int m = chunk >> 3, kk8 = (chunk & 7) * 8;
    async_cp16(&As[m][kk8], &xsrc[(bb * TM + m) * HID + kl + kk8]);
  }
#pragma unroll
  for (int it = 0; it < 2; ++it) {
    int chunk = tid + it * 256;
    int n = chunk >> 3, kk8 = (chunk & 7) * 8;
    async_cp16(&Bs[n][kk8], &W[(nb * TN + n) * K + kc0 + kk8]);
  }
}

// ---------------------------------------------------------------------------
// Fused LSTM step: gates GEMM (bf16 WMMA) + bias + rank-1 input + cell update.
//   x0 : [1024][256] bf16 (K rows 0..255)   x1 : [1024][256] bf16 (K 256..511)
//   W  : [1024][K]   bf16, N-major, gate-interleaved
//   xvec (optional): per-batch scalar input, gates += xvec[b]*wih0P[n']
//   cBuf: [1024][256] f32 in-place ; hOut: [1024][256] bf16 (next step operand)
// Grid (8,16): blockIdx.x = batch block of 128, blockIdx.y = n' block of 64.
// ---------------------------------------------------------------------------
__global__ __launch_bounds__(256)
void lstm_step(const __bf16* __restrict__ x0,
               const __bf16* __restrict__ x1,
               const __bf16* __restrict__ W,
               const float*  __restrict__ biasP,
               const float*  __restrict__ wih0P,
               const float*  __restrict__ xvec, int xstride, int xoff,
               float*  __restrict__ cBuf,
               __bf16* __restrict__ hOut,
               int K) {
  // Double-buffered A/B panels, C spill overlaid (union) for the epilogue.
  struct SMab {
    __bf16 As[2][TM][72];   // [m][k], 144B rows (16B aligned)
    __bf16 Bs[2][TN][80];   // transposed [n][k], 160B rows (32B aligned)
  };
  __shared__ union {
    SMab ab;
    float Cs[TM][68];       // 272B rows (16B aligned)
  } sm;

  const int tid  = threadIdx.x;
  const int bb   = blockIdx.x;
  const int nb   = blockIdx.y;
  const int lane = tid & 31;
  const int wave = tid >> 5;           // 8 waves
  const int wm   = (wave & 3) * 32;    // 4 waves along M
  const int wn   = (wave >> 2) * 32;   // 2 waves along N
  const int half = lane >> 4;
  const int l16  = lane & 15;

  v8f acc[2][2];
#pragma unroll
  for (int i = 0; i < 2; ++i)
#pragma unroll
    for (int j = 0; j < 2; ++j)
      acc[i][j] = (v8f){0,0,0,0,0,0,0,0};

  const int niter = K / TK;            // 4 (layer0) or 8 (layer1)
  stage_chunk(x0, x1, W, bb, nb, K, 0, tid, sm.ab.As[0], sm.ab.Bs[0]);

  for (int i = 0; i < niter; ++i) {
    wait_async0();        // my async copies for buffer i&1 have landed
    __syncthreads();      // everyone's copies landed; everyone done with i-1
    if (i + 1 < niter)    // prefetch next chunk into the other buffer
      stage_chunk(x0, x1, W, bb, nb, K, (i + 1) * TK, tid,
                  sm.ab.As[(i + 1) & 1], sm.ab.Bs[(i + 1) & 1]);

    const __bf16 (*As)[72] = sm.ab.As[i & 1];
    const __bf16 (*Bs)[80] = sm.ab.Bs[i & 1];

#pragma unroll
    for (int kc = 0; kc < TK; kc += 32) {
      // A fragment: lane(m=l16) gets K {16h..16h+15} contiguous (matches B)
      union { v16bf v; uint4 q[2]; } af[2];
#pragma unroll
      for (int i2 = 0; i2 < 2; ++i2) {
        const __bf16* rowp = &As[wm + i2 * 16 + l16][0];
        af[i2].q[0] = *(const uint4*)&rowp[kc + 16 * half];
        af[i2].q[1] = *(const uint4*)&rowp[kc + 16 * half + 8];
      }
      // B fragment: lane(n=l16) gets K {16h..16h+15} contiguous (Bs is [n][k])
      v16bf bf[2];
#pragma unroll
      for (int j = 0; j < 2; ++j)
        bf[j] = *(const v16bf*)&Bs[wn + j * 16 + l16][kc + 16 * half];

#pragma unroll
      for (int i2 = 0; i2 < 2; ++i2)
#pragma unroll
        for (int j = 0; j < 2; ++j)
          acc[i2][j] = __builtin_amdgcn_wmma_f32_16x16x32_bf16(
              false, af[i2].v, false, bf[j], (short)0, acc[i2][j], false, false);
    }
  }

  __syncthreads();   // all waves done reading A/B panels (union reuse below)

  // ---- spill accumulators: D layout row = r + 8*half, col = l16 ----
#pragma unroll
  for (int i = 0; i < 2; ++i)
#pragma unroll
    for (int j = 0; j < 2; ++j)
#pragma unroll
      for (int r = 0; r < 8; ++r)
        sm.Cs[wm + i * 16 + r + 8 * half][wn + j * 16 + l16] = acc[i][j][r];
  __syncthreads();

  // ---- fused LSTM cell: 128 batch x 16 units, 4 adjacent gate columns ----
  for (int s = 0; s < 8; ++s) {
    int pt = tid + s * 256;
    int b = pt >> 4, u = pt & 15;
    float4 g4 = *(float4*)&sm.Cs[b][u * 4];
    int ng = nb * TN + u * 4;
    const float4 b4 = *(const float4*)&biasP[ng];
    g4.x += b4.x; g4.y += b4.y; g4.z += b4.z; g4.w += b4.w;
    int B_ = bb * TM + b;
    if (xvec) {
      float xb = xvec[B_ * xstride + xoff];
      const float4 w4 = *(const float4*)&wih0P[ng];
      g4.x += xb * w4.x; g4.y += xb * w4.y; g4.z += xb * w4.z; g4.w += xb * w4.w;
    }
    float gi = 1.0f / (1.0f + __expf(-g4.x));   // input gate
    float gf = 1.0f / (1.0f + __expf(-g4.y));   // forget gate
    float gg = tanhf(g4.z);                      // cell candidate
    float go = 1.0f / (1.0f + __expf(-g4.w));   // output gate
    int U = nb * 16 + u;
    float cOld = cBuf[B_ * HID + U];
    float cNew = gf * cOld + gi * gg;
    cBuf[B_ * HID + U] = cNew;
    hOut[B_ * HID + U] = (__bf16)(go * tanhf(cNew));
  }
}

// ---------------------------------------------------------------------------
extern "C" void kernel_launch(void* const* d_in, const int* in_sizes, int n_in,
                              void* d_out, int out_size, void* d_ws, size_t ws_size,
                              hipStream_t stream) {
  const float* x     = (const float*)d_in[0];
  const float* eWih0 = (const float*)d_in[1];
  const float* eWhh0 = (const float*)d_in[2];
  const float* eb0   = (const float*)d_in[3];
  const float* eWih1 = (const float*)d_in[4];
  const float* eWhh1 = (const float*)d_in[5];
  const float* eb1   = (const float*)d_in[6];
  const float* dWih0 = (const float*)d_in[7];
  const float* dWhh0 = (const float*)d_in[8];
  const float* db0   = (const float*)d_in[9];
  const float* dWih1 = (const float*)d_in[10];
  const float* dWhh1 = (const float*)d_in[11];
  const float* db1   = (const float*)d_in[12];
  const float* fcW   = (const float*)d_in[13];
  const float* fcb   = (const float*)d_in[14];
  float* out = (float*)d_out;

  // ---- carve workspace (all regions 256B aligned) ----
  size_t off = 0;
  auto carve = [&](size_t bytes) -> char* {
    char* p = (char*)d_ws + off;
    off += (bytes + 255) & ~(size_t)255;
    return p;
  };
  __bf16* eW0p = (__bf16*)carve((size_t)G4 * HID * 2);       // enc L0 (Whh only)
  __bf16* eW1p = (__bf16*)carve((size_t)G4 * 2 * HID * 2);   // enc L1 [Wih|Whh]
  __bf16* dW0p = (__bf16*)carve((size_t)G4 * HID * 2);       // dec L0
  __bf16* dW1p = (__bf16*)carve((size_t)G4 * 2 * HID * 2);   // dec L1
  float* eb0p   = (float*)carve(G4 * 4);
  float* eb1p   = (float*)carve(G4 * 4);
  float* db0p   = (float*)carve(G4 * 4);
  float* db1p   = (float*)carve(G4 * 4);
  float* ewih0p = (float*)carve(G4 * 4);
  float* dwih0p = (float*)carve(G4 * 4);
  __bf16* h0buf[2], *h1buf[2];
  h0buf[0] = (__bf16*)carve((size_t)B_TOT * HID * 2);
  h0buf[1] = (__bf16*)carve((size_t)B_TOT * HID * 2);
  h1buf[0] = (__bf16*)carve((size_t)B_TOT * HID * 2);
  h1buf[1] = (__bf16*)carve((size_t)B_TOT * HID * 2);
  float* c0 = (float*)carve((size_t)B_TOT * HID * 4);
  float* c1 = (float*)carve((size_t)B_TOT * HID * 4);
  float* predbuf = (float*)carve(B_TOT * 4);
  (void)ws_size; (void)in_sizes; (void)n_in; (void)out_size;

  // ---- weight / bias prepack ----
  pack_weights<<<(G4 * HID + 255) / 256, 256, 0, stream>>>(eW0p, nullptr, eWhh0, 0, HID);
  pack_weights<<<(G4 * 2 * HID + 255) / 256, 256, 0, stream>>>(eW1p, eWih1, eWhh1, HID, 2 * HID);
  pack_weights<<<(G4 * HID + 255) / 256, 256, 0, stream>>>(dW0p, nullptr, dWhh0, 0, HID);
  pack_weights<<<(G4 * 2 * HID + 255) / 256, 256, 0, stream>>>(dW1p, dWih1, dWhh1, HID, 2 * HID);
  pack_vec<<<4, 256, 0, stream>>>(eb0p, eb0);
  pack_vec<<<4, 256, 0, stream>>>(eb1p, eb1);
  pack_vec<<<4, 256, 0, stream>>>(db0p, db0);
  pack_vec<<<4, 256, 0, stream>>>(db1p, db1);
  pack_vec<<<4, 256, 0, stream>>>(ewih0p, eWih0);   // (4H,1) -> gate-interleaved
  pack_vec<<<4, 256, 0, stream>>>(dwih0p, dWih0);
  zero_state<<<(B_TOT * HID + 255) / 256, 256, 0, stream>>>(h0buf[0], h1buf[0], c0, c1);

  const dim3 gGrid(B_TOT / TM, G4 / TN);   // (8,16)

  // ---- encoder: 336 steps, 2 layers (layer-0 rank-1 input from x column t) ----
  int p = 0, q = 0;
  for (int t = 0; t < LOOKB; ++t) {
    lstm_step<<<gGrid, 256, 0, stream>>>(h0buf[p], nullptr, eW0p, eb0p, ewih0p,
                                         x, LOOKB, t, c0, h0buf[1 - p], HID);
    lstm_step<<<gGrid, 256, 0, stream>>>(h0buf[1 - p], h1buf[q], eW1p, eb1p,
                                         nullptr, nullptr, 0, 0, c1, h1buf[1 - q],
                                         2 * HID);
    p ^= 1; q ^= 1;
  }

  // ---- autoregressive decoder: 168 steps ----
  seed_pred<<<4, 256, 0, stream>>>(predbuf, x);
  for (int t = 0; t < HORIZ; ++t) {
    lstm_step<<<gGrid, 256, 0, stream>>>(h0buf[p], nullptr, dW0p, db0p, dwih0p,
                                         predbuf, 1, 0, c0, h0buf[1 - p], HID);
    lstm_step<<<gGrid, 256, 0, stream>>>(h0buf[1 - p], h1buf[q], dW1p, db1p,
                                         nullptr, nullptr, 0, 0, c1, h1buf[1 - q],
                                         2 * HID);
    fc_pred<<<4, 256, 0, stream>>>(h1buf[1 - q], fcW, fcb, predbuf, out, t);
    p ^= 1; q ^= 1;
  }
}
